// RelationInnerProdSelfAttention_69827578298377
// MI455X (gfx1250) — matrix operand order repaired and checked by reading, not compiled, not run on hardware
//
#include <hip/hip_runtime.h>
#include <hip/hip_bf16.h>

// ---------------- problem geometry (BERT-base relational attention) --------
constexpr int Bq   = 4;
constexpr int Nn   = 512;
constexpr int Hh   = 12;
constexpr int Dh   = 64;
constexpr int HID  = Hh * Dh;      // 768
constexpr int Ee   = 65536;
constexpr int NREL = 64;
constexpr int MROW = Bq * Nn;      // 2048 rows of X/Q/K/V
constexpr int NSEG = Bq * Nn;      // softmax segments
// ---------------------------------------------------------------------------

typedef __attribute__((ext_vector_type(16))) _Float16 v16h;
typedef __attribute__((ext_vector_type(8)))  _Float16 v8h;
typedef __attribute__((ext_vector_type(8)))  float    v8f;

// order-preserving int encoding of float for atomicMax (non-NaN)
__device__ __forceinline__ int fenc(float f) {
    int i = __float_as_int(f);
    return (i >= 0) ? i : (i ^ 0x7FFFFFFF);
}
__device__ __forceinline__ float fdec(int i) {
    return __int_as_float((i >= 0) ? i : (i ^ 0x7FFFFFFF));
}

// -------------------- small prep kernels -----------------------------------
__global__ __launch_bounds__(256) void cvt_f32_to_f16_kernel(const float* __restrict__ src,
                                                             _Float16* __restrict__ dst, int n) {
    int i = blockIdx.x * 256 + threadIdx.x;
    if (i < n) dst[i] = (_Float16)src[i];
}

// ReT[r][n][k] = rel[r][k][n]  (so WMMA B-fragments load contiguous k)
__global__ __launch_bounds__(256) void rel_transpose_kernel(const float* __restrict__ rel,
                                                            _Float16* __restrict__ ReT) {
    int i = blockIdx.x * 256 + threadIdx.x;          // NREL*64*64
    if (i >= NREL * Dh * Dh) return;
    int r = i >> 12;
    int n = (i >> 6) & 63;
    int k = i & 63;
    ReT[i] = (_Float16)rel[(r << 12) + (k << 6) + n];
}

__global__ __launch_bounds__(256) void init_kernel(float* __restrict__ out,
                                                   int* __restrict__ maxi,
                                                   float* __restrict__ denom,
                                                   int nout, int nseg) {
    int i = blockIdx.x * 256 + threadIdx.x;
    if (i < nout) out[i] = 0.0f;
    if (i < nseg) { maxi[i] = (int)0x80000000; denom[i] = 0.0f; }
}

// -------------------- phase 1: QKV projection via WMMA ---------------------
// Out[m,n] = sum_k X[m,k] * W[n,k] + bias[n]   (torch Linear: y = x @ W^T + b)
// One wave computes a 16x64 output tile: one A fragment per k-step reused
// across 4 B fragments / 4 accumulators (4 independent WMMAs per k-step).
__global__ __launch_bounds__(32) void qkv_gemm_kernel(const _Float16* __restrict__ Xh,
                                                      const _Float16* __restrict__ Wh3,
                                                      const float* __restrict__ bqv,
                                                      const float* __restrict__ bkv,
                                                      const float* __restrict__ bvv,
                                                      _Float16* __restrict__ Qh,
                                                      float* __restrict__ Kf,
                                                      float* __restrict__ Vf) {
    const int lane = threadIdx.x & 31;
    const int m0 = blockIdx.x * 16;
    const int n0 = blockIdx.y * 64;
    const int z  = blockIdx.z;                     // 0=Q,1=K,2=V
    const _Float16* W = Wh3 + (size_t)z * HID * HID;

    const int kh = (lane >> 4) * 16;               // A layout: contiguous 16 halves
    const int kb = (lane >> 4) * 16;               // B layout: contiguous 16 halves
    const int mr = m0 + (lane & 15);
    const int nlo = lane & 15;

    const _Float16* arow = Xh + (size_t)mr * HID + kh;
    const _Float16* brow0 = W + (size_t)(n0 + 0 * 16 + nlo) * HID + kb;
    const _Float16* brow1 = W + (size_t)(n0 + 1 * 16 + nlo) * HID + kb;
    const _Float16* brow2 = W + (size_t)(n0 + 2 * 16 + nlo) * HID + kb;
    const _Float16* brow3 = W + (size_t)(n0 + 3 * 16 + nlo) * HID + kb;

    v8f acc0 = {}, acc1 = {}, acc2 = {}, acc3 = {};
    for (int k0 = 0; k0 < HID; k0 += 32) {
        union { v16h v; v8h h[2]; } ua;
        ua.h[0] = *(const v8h*)(arow + k0);
        ua.h[1] = *(const v8h*)(arow + k0 + 8);
        v16h b0 = *(const v16h*)(brow0 + k0);
        v16h b1 = *(const v16h*)(brow1 + k0);
        v16h b2 = *(const v16h*)(brow2 + k0);
        v16h b3 = *(const v16h*)(brow3 + k0);
        acc0 = __builtin_amdgcn_wmma_f32_16x16x32_f16(false, ua.v, false, b0,
                                                      (short)0, acc0, false, false);
        acc1 = __builtin_amdgcn_wmma_f32_16x16x32_f16(false, ua.v, false, b1,
                                                      (short)0, acc1, false, false);
        acc2 = __builtin_amdgcn_wmma_f32_16x16x32_f16(false, ua.v, false, b2,
                                                      (short)0, acc2, false, false);
        acc3 = __builtin_amdgcn_wmma_f32_16x16x32_f16(false, ua.v, false, b3,
                                                      (short)0, acc3, false, false);
    }

    const float* bias = (z == 0) ? bqv : ((z == 1) ? bkv : bvv);
    const int mAcc = (lane >> 4) << 3;
#pragma unroll
    for (int j = 0; j < 4; ++j) {
        const v8f& acc = (j == 0) ? acc0 : ((j == 1) ? acc1 : ((j == 2) ? acc2 : acc3));
        const int col = n0 + j * 16 + nlo;
        const float bcol = bias[col];
#pragma unroll
        for (int r = 0; r < 8; ++r) {
            const int row = m0 + r + mAcc;
            const size_t idx = (size_t)row * HID + col;
            const float val = acc[r] + bcol;
            if (z == 0)      Qh[idx] = (_Float16)val;
            else if (z == 1) Kf[idx] = val;
            else             Vf[idx] = val;
        }
    }
}

// -------------------- phase 2: per-edge relation logits via WMMA -----------
// Per edge: Qp(16x64, heads padded 12->16) = Qe(16x64) @ Re(64x64),
// logit[h] = dot(Qp[h,:], Ke[h,:]) / 8.  One wave per edge, 8 WMMAs.
__global__ __launch_bounds__(256) void edge_logits_kernel(const int* __restrict__ ei,
                                                          const _Float16* __restrict__ Qh,
                                                          const float* __restrict__ Kf,
                                                          const _Float16* __restrict__ ReT,
                                                          float* __restrict__ logits,
                                                          int* __restrict__ maxi) {
    const int lane = threadIdx.x & 31;
    const int wave = threadIdx.x >> 5;
    const int e = blockIdx.x * 8 + wave;

    const int be = ei[e];
    const int he = ei[Ee + e];
    const int te = ei[2 * Ee + e];
    const int re = ei[3 * Ee + e];

    const _Float16* Qrow = Qh + (size_t)(be * Nn + he) * HID;
    const float*    Krow = Kf + (size_t)(be * Nn + te) * HID;
    const _Float16* RT   = ReT + (size_t)re * (Dh * Dh);

    const int mrow = lane & 15;                // A row (head index) loaded by this lane
    const int kh   = (lane >> 4) * 16;
    const int kb   = (lane >> 4) * 16;
    const int nlo  = lane & 15;

    // A fragments for k-steps 0 and 32 (heads >= 12 are zero padding)
    v16h a0 = {}, a1 = {};
    if (mrow < Hh) {
        union { v16h v; v8h h[2]; } u0, u1;
        const _Float16* q = Qrow + mrow * Dh + kh;
        u0.h[0] = *(const v8h*)(q);
        u0.h[1] = *(const v8h*)(q + 8);
        u1.h[0] = *(const v8h*)(q + 32);
        u1.h[1] = *(const v8h*)(q + 40);
        a0 = u0.v;
        a1 = u1.v;
    }

    float part[8];
#pragma unroll
    for (int r = 0; r < 8; ++r) part[r] = 0.0f;

    const int mAcc = (lane >> 4) << 3;
#pragma unroll
    for (int nt = 0; nt < 4; ++nt) {
        const int n = nt * 16 + nlo;
        v16h b0 = *(const v16h*)(RT + (size_t)n * Dh + kb);        // k = 0..31
        v16h b1 = *(const v16h*)(RT + (size_t)n * Dh + 32 + kb);   // k = 32..63
        v8f acc = {};
        acc = __builtin_amdgcn_wmma_f32_16x16x32_f16(false, a0, false, b0,
                                                     (short)0, acc, false, false);
        acc = __builtin_amdgcn_wmma_f32_16x16x32_f16(false, a1, false, b1,
                                                     (short)0, acc, false, false);
#pragma unroll
        for (int r = 0; r < 8; ++r) {
            const int m = r + mAcc;                           // head of acc element
            const float kv = (m < Hh) ? Krow[m * Dh + n] : 0.0f;
            part[r] += acc[r] * kv;
        }
    }

    // reduce over N (16 lanes per half; halves hold different heads)
#pragma unroll
    for (int r = 0; r < 8; ++r) {
        float v = part[r];
        v += __shfl_xor(v, 1);
        v += __shfl_xor(v, 2);
        v += __shfl_xor(v, 4);
        v += __shfl_xor(v, 8);
        part[r] = v;
    }

    const float scale = 0.125f;                               // 1/sqrt(64)
    const int segbase = (be * Nn + he) * Hh;
    if (lane == 0) {
#pragma unroll
        for (int r = 0; r < 8; ++r) {                         // heads 0..7
            const float lg = part[r] * scale;
            logits[(size_t)e * Hh + r] = lg;
            atomicMax(&maxi[segbase + r], fenc(lg));
        }
    } else if (lane == 16) {
#pragma unroll
        for (int r = 0; r < 4; ++r) {                         // heads 8..11
            const float lg = part[r] * scale;
            logits[(size_t)e * Hh + 8 + r] = lg;
            atomicMax(&maxi[segbase + 8 + r], fenc(lg));
        }
    }
}

// -------------------- phase 3: exp + segment denominators ------------------
__global__ __launch_bounds__(256) void softmax_denom_kernel(const int* __restrict__ ei,
                                                            float* __restrict__ logits,
                                                            const int* __restrict__ maxi,
                                                            float* __restrict__ denom) {
    int t = blockIdx.x * 256 + threadIdx.x;                   // Ee*Hh
    if (t >= Ee * Hh) return;
    int e = t / Hh;
    int h = t - e * Hh;
    int be = ei[e];
    int he = ei[Ee + e];
    int seg = be * Nn + he;
    float m = fdec(maxi[seg * Hh + h]);
    float ex = __expf(logits[t] - m);
    logits[t] = ex;                                           // store ex in place
    atomicAdd(&denom[seg * Hh + h], ex);
}

// -------------------- phase 4: weighted aggregation of V[tail] -------------
__global__ __launch_bounds__(256) void aggregate_kernel(const int* __restrict__ ei,
                                                        const float* __restrict__ ex,
                                                        const float* __restrict__ denom,
                                                        const float* __restrict__ Vf,
                                                        float* __restrict__ out) {
    int t = blockIdx.x * 256 + threadIdx.x;                   // Ee*Hh*16 (float4 chunks)
    if (t >= Ee * Hh * (Dh / 4)) return;
    int e   = t / (Hh * (Dh / 4));
    int rem = t - e * (Hh * (Dh / 4));
    int h   = rem >> 4;
    int d4  = rem & 15;

    int be = ei[e];
    int he = ei[Ee + e];
    int te = ei[2 * Ee + e];
    int seg = be * Nn + he;

    float p = ex[(size_t)e * Hh + h] / denom[seg * Hh + h];
    const float4 v = *(const float4*)(Vf + (size_t)(be * Nn + te) * HID + h * Dh + d4 * 4);
    float* o = out + (size_t)seg * HID + h * Dh + d4 * 4;
    atomicAdd(o + 0, p * v.x);
    atomicAdd(o + 1, p * v.y);
    atomicAdd(o + 2, p * v.z);
    atomicAdd(o + 3, p * v.w);
}

// ---------------------------------------------------------------------------
extern "C" void kernel_launch(void* const* d_in, const int* in_sizes, int n_in,
                              void* d_out, int out_size, void* d_ws, size_t ws_size,
                              hipStream_t stream) {
    const float* X    = (const float*)d_in[0];   // (B,N,768)
    const int*   ei   = (const int*)  d_in[1];   // (4,E)
    const float* Wq   = (const float*)d_in[3];
    const float* bq   = (const float*)d_in[4];
    const float* Wk   = (const float*)d_in[5];
    const float* bk   = (const float*)d_in[6];
    const float* Wv   = (const float*)d_in[7];
    const float* bv   = (const float*)d_in[8];
    const float* rel  = (const float*)d_in[9];   // (NREL,64,64)
    float* out = (float*)d_out;

    // workspace carve-out (~27 MB)
    char* w = (char*)d_ws;
    auto carve = [&](size_t bytes) -> void* {
        void* p = (void*)w;
        w += (bytes + 255) & ~(size_t)255;
        return p;
    };
    _Float16* Xh    = (_Float16*)carve((size_t)MROW * HID * 2);
    _Float16* Wh3   = (_Float16*)carve((size_t)3 * HID * HID * 2);
    _Float16* ReT   = (_Float16*)carve((size_t)NREL * Dh * Dh * 2);
    _Float16* Qh    = (_Float16*)carve((size_t)MROW * HID * 2);
    float*    Kf    = (float*)   carve((size_t)MROW * HID * 4);
    float*    Vf    = (float*)   carve((size_t)MROW * HID * 4);
    float*    logit = (float*)   carve((size_t)Ee * Hh * 4);
    int*      maxi  = (int*)     carve((size_t)NSEG * Hh * 4);
    float*    denom = (float*)   carve((size_t)NSEG * Hh * 4);

    const int nX = MROW * HID;          // 1,572,864
    const int nW = HID * HID;           // 589,824

    // phase 0: precision conversion + transposed relation matrices + init
    cvt_f32_to_f16_kernel<<<(nX + 255) / 256, 256, 0, stream>>>(X, Xh, nX);
    cvt_f32_to_f16_kernel<<<(nW + 255) / 256, 256, 0, stream>>>(Wq, Wh3 + 0 * (size_t)nW, nW);
    cvt_f32_to_f16_kernel<<<(nW + 255) / 256, 256, 0, stream>>>(Wk, Wh3 + 1 * (size_t)nW, nW);
    cvt_f32_to_f16_kernel<<<(nW + 255) / 256, 256, 0, stream>>>(Wv, Wh3 + 2 * (size_t)nW, nW);
    rel_transpose_kernel<<<(NREL * Dh * Dh + 255) / 256, 256, 0, stream>>>(rel, ReT);
    init_kernel<<<(nX + 255) / 256, 256, 0, stream>>>(out, maxi, denom, nX, NSEG * Hh);

    // phase 1: Q/K/V GEMMs (WMMA), one wave per 16x64 tile (4-way N blocking)
    qkv_gemm_kernel<<<dim3(MROW / 16, HID / 64, 3), 32, 0, stream>>>(
        Xh, Wh3, bq, bk, bv, Qh, Kf, Vf);

    // phase 2: per-edge relation-projected logits (WMMA), one wave per edge
    edge_logits_kernel<<<Ee / 8, 256, 0, stream>>>(ei, Qh, Kf, ReT, logit, maxi);

    // phase 3: segment softmax denominators
    softmax_denom_kernel<<<(Ee * Hh + 255) / 256, 256, 0, stream>>>(ei, logit, maxi, denom);

    // phase 4: weighted aggregation into output
    aggregate_kernel<<<(Ee * Hh * (Dh / 4) + 255) / 256, 256, 0, stream>>>(
        ei, logit, denom, Vf, out);
}
